// BahdanauAttention_72713796321551
// MI455X (gfx1250) — matrix operand order repaired and checked by reading, not compiled, not run on hardware
//
#include <hip/hip_runtime.h>
#include <hip/hip_bf16.h>
#include <math.h>

typedef __attribute__((ext_vector_type(16))) __bf16 v16bf;
typedef __attribute__((ext_vector_type(8)))  __bf16 v8bf;
typedef __attribute__((ext_vector_type(8)))  float  v8f;
typedef __attribute__((ext_vector_type(4)))  float  v4f;

#define HDIM 2048        // K of the per-row GEMM segment and N (out features)
#define TM   64          // rows per workgroup
#define LDSS 2056        // LDS row stride in bf16 elements (2048 + 8 pad, keeps 16B align)

static __device__ __forceinline__ unsigned short f32_to_bf16(float f) {
    union { float f; unsigned u; } v; v.f = f;
    unsigned u = v.u;
    u += 0x7FFFu + ((u >> 16) & 1u);   // round-to-nearest-even
    return (unsigned short)(u >> 16);
}

// ---------------------------------------------------------------------------
// Convert first 2048 columns of W (row stride wld) to bf16, row stride 2048.
// ---------------------------------------------------------------------------
__global__ void wconv_kernel(const float* __restrict__ W,
                             unsigned short* __restrict__ Wb, int wld) {
    int n = blockIdx.x;                       // one output row per block
    const float* src = W + (size_t)n * wld;
    unsigned short* dst = Wb + (size_t)n * HDIM;
    for (int k = threadIdx.x; k < HDIM; k += 256) dst[k] = f32_to_bf16(src[k]);
}

// ---------------------------------------------------------------------------
// Per-batch rank terms: bc[b,n] = bias[n] + q[b]*W[n,qbase:..] (+ c[b]*W[n,cbase:..])
// ---------------------------------------------------------------------------
__global__ void bc_kernel(const float* __restrict__ W, int wld,
                          const float* __restrict__ bias,
                          const float* __restrict__ q, int qlen, int qbase,
                          const float* __restrict__ c, int clen, int cbase,
                          float* __restrict__ bc) {
    int b = blockIdx.x;
    int n = blockIdx.y * 256 + threadIdx.x;
    const float* wrow = W + (size_t)n * wld;
    float acc = bias[n];
    const float* qb = q + (size_t)b * qlen;
    for (int j = 0; j < qlen; ++j) acc += qb[j] * wrow[qbase + j];
    if (c) {
        const float* cb = c + (size_t)b * clen;
        for (int j = 0; j < clen; ++j) acc += cb[j] * wrow[cbase + j];
    }
    bc[(size_t)b * HDIM + n] = acc;
}

// ---------------------------------------------------------------------------
// Fused GEMM + tanh + dot(other) -> scores[row].
// Each WG: 64 rows staged once into LDS as bf16; 8 waves sweep all 2048 cols,
// each wave: 4 row-tiles of v_wmma_f32_16x16x32_bf16 per K-step, fragments
// double-buffered so DS/VMEM latency is covered by WMMA issue.
// ---------------------------------------------------------------------------
__global__ __launch_bounds__(256)
void gemm_score_kernel(const float* __restrict__ A,
                       const unsigned short* __restrict__ WbU,
                       const float* __restrict__ Wfull, int wld,
                       const float* __restrict__ bc,
                       const float* __restrict__ state,
                       const float* __restrict__ other,
                       float* __restrict__ scores,
                       int L) {
    extern __shared__ char smem[];
    unsigned short* sA = (unsigned short*)smem;                  // TM x LDSS bf16
    float* sScore = (float*)(smem + (size_t)TM * LDSS * 2);      // TM
    float* sState = sScore + TM;                                 // TM

    const int tid  = threadIdx.x;
    const int wave = tid >> 5;
    const int ln   = tid & 31;
    const int rowBase = blockIdx.x * TM;
    const int b = rowBase / L;                                   // 64 | L for both sides

    // ---- stage 64x2048 feat strip, f32 -> bf16 (read from HBM exactly once) ----
    {
        const float* src = A + (size_t)rowBase * HDIM;
        for (int i = tid; i < (TM * HDIM / 4); i += 256) {       // 128 float4 per thread
            int row = i >> 9;                                    // 512 float4 per row
            int c4  = i & 511;
            v4f v = *(const v4f*)(src + (size_t)row * HDIM + c4 * 4);
            unsigned short* d = sA + (size_t)row * LDSS + c4 * 4;
            d[0] = f32_to_bf16(v.x); d[1] = f32_to_bf16(v.y);
            d[2] = f32_to_bf16(v.z); d[3] = f32_to_bf16(v.w);
        }
        if (tid < TM) { sScore[tid] = 0.0f; sState[tid] = state[rowBase + tid]; }
    }
    __syncthreads();

    const int half = ln >> 4;        // 0: lanes 0-15, 1: lanes 16-31
    const int l16  = ln & 15;

    // Per-rt LDS base pointers (kk*2 bytes then folds into the DS offset imm).
    const __bf16* aptr[4];
    #pragma unroll
    for (int rt = 0; rt < 4; ++rt)
        aptr[rt] = (const __bf16*)(sA + (size_t)(rt * 16 + l16) * LDSS + half * 16);

    // Per-lane deferred score partials: sacc[rt][i] accumulates over all columns.
    float sacc[4][8];
    #pragma unroll
    for (int rt = 0; rt < 4; ++rt)
        #pragma unroll
        for (int i = 0; i < 8; ++i) sacc[rt][i] = 0.0f;

    for (int cbk = 0; cbk < HDIM / 128; ++cbk) {
        const int n = cbk * 128 + wave * 16 + l16;               // this lane's column
        v8f acc[4] = { {}, {}, {}, {} };
        const __bf16* wrow = (const __bf16*)(WbU + (size_t)n * HDIM) + half * 16;

        // ---- prologue: fragments for kk = 0 ----
        v8bf blo = *(const v8bf*)(wrow);
        v8bf bhi = *(const v8bf*)(wrow + 8);
        v8bf alo[4], ahi[4];
        #pragma unroll
        for (int rt = 0; rt < 4; ++rt) {
            alo[rt] = *(const v8bf*)(aptr[rt]);
            ahi[rt] = *(const v8bf*)(aptr[rt] + 8);
        }

        #pragma unroll 2
        for (int kk = 0; kk < HDIM; kk += 32) {
            // ---- issue next K-step's loads before consuming current frags ----
            const int kn = (kk + 32 < HDIM) ? kk + 32 : 0;       // uniform wrap
            v8bf nblo = *(const v8bf*)(wrow + kn);
            v8bf nbhi = *(const v8bf*)(wrow + kn + 8);
            v8bf nalo[4], nahi[4];
            #pragma unroll
            for (int rt = 0; rt < 4; ++rt) {
                nalo[rt] = *(const v8bf*)(aptr[rt] + kn);
                nahi[rt] = *(const v8bf*)(aptr[rt] + kn + 8);
            }
            // ---- 4 WMMAs on the current fragments ----
            v16bf bf = __builtin_shufflevector(blo, bhi,
                0,1,2,3,4,5,6,7,8,9,10,11,12,13,14,15);
            #pragma unroll
            for (int rt = 0; rt < 4; ++rt) {
                v16bf af = __builtin_shufflevector(alo[rt], ahi[rt],
                    0,1,2,3,4,5,6,7,8,9,10,11,12,13,14,15);
                acc[rt] = __builtin_amdgcn_wmma_f32_16x16x32_bf16(
                    false, af, false, bf, (short)0, acc[rt], false, false);
            }
            // ---- rotate buffers ----
            blo = nblo; bhi = nbhi;
            #pragma unroll
            for (int rt = 0; rt < 4; ++rt) { alo[rt] = nalo[rt]; ahi[rt] = nahi[rt]; }
        }

        // ---- per-column epilogue: + bc + state*w3072, tanh, * other[n] ----
        const float wc  = Wfull[(size_t)n * wld + 3072];
        const float bcv = bc[(size_t)b * HDIM + n];
        const float ov  = other[n];
        #pragma unroll
        for (int rt = 0; rt < 4; ++rt) {
            #pragma unroll
            for (int i = 0; i < 8; ++i) {
                int M = rt * 16 + half * 8 + i;                  // row within WG tile
                float h = tanhf(acc[rt][i] + bcv + sState[M] * wc);
                sacc[rt][i] += h * ov;                           // defer reduction
            }
        }
    }

    // ---- one cross-lane reduction + LDS accumulate at the very end ----
    #pragma unroll
    for (int rt = 0; rt < 4; ++rt) {
        #pragma unroll
        for (int i = 0; i < 8; ++i) {
            float s = sacc[rt][i];
            s += __shfl_xor(s, 1, 16);
            s += __shfl_xor(s, 2, 16);
            s += __shfl_xor(s, 4, 16);
            s += __shfl_xor(s, 8, 16);
            if (l16 == 0)
                atomicAdd(&sScore[rt * 16 + half * 8 + i], s);   // ds_add_f32
        }
    }
    __syncthreads();
    if (tid < TM) scores[rowBase + tid] = sScore[tid];
}

// ---------------------------------------------------------------------------
// Masked softmax over L per batch.
// ---------------------------------------------------------------------------
__global__ void softmax_kernel(const float* __restrict__ scores,
                               const int* __restrict__ mask,
                               float* __restrict__ alphas, int L) {
    int b = blockIdx.x;
    __shared__ float red[256];
    float m = -__builtin_inff();
    for (int l = threadIdx.x; l < L; l += 256) {
        float s = (mask[b * L + l] == 0) ? -__builtin_inff() : scores[b * L + l];
        m = fmaxf(m, s);
    }
    red[threadIdx.x] = m; __syncthreads();
    for (int off = 128; off; off >>= 1) {
        if (threadIdx.x < off) red[threadIdx.x] = fmaxf(red[threadIdx.x], red[threadIdx.x + off]);
        __syncthreads();
    }
    m = red[0]; __syncthreads();
    float sum = 0.0f;
    for (int l = threadIdx.x; l < L; l += 256) {
        float e = (mask[b * L + l] == 0) ? 0.0f : __expf(scores[b * L + l] - m);
        alphas[b * L + l] = e;
        sum += e;
    }
    red[threadIdx.x] = sum; __syncthreads();
    for (int off = 128; off; off >>= 1) {
        if (threadIdx.x < off) red[threadIdx.x] += red[threadIdx.x + off];
        __syncthreads();
    }
    float inv = 1.0f / red[0];
    for (int l = threadIdx.x; l < L; l += 256) alphas[b * L + l] *= inv;
}

// ---------------------------------------------------------------------------
// ctx[b,d] = sum_l alphas[b,l] * V[b,l,d]  (D = 2048)
// ---------------------------------------------------------------------------
__global__ void context_kernel(const float* __restrict__ alphas,
                               const float* __restrict__ V,
                               float* __restrict__ ctx, int L) {
    int b = blockIdx.x;
    int d = blockIdx.y * 256 + threadIdx.x;
    const float* vb = V + (size_t)b * L * HDIM;
    const float* ab = alphas + (size_t)b * L;
    float acc = 0.0f;
    for (int l = 0; l < L; ++l) acc += ab[l] * vb[(size_t)l * HDIM + d];
    ctx[(size_t)b * HDIM + d] = acc;
}

// ---------------------------------------------------------------------------
// loss = sum_l min( min_b state[b,l], min_b alphas[b,l] )
// ---------------------------------------------------------------------------
__global__ void loss_kernel(const float* __restrict__ state,
                            const float* __restrict__ alphas,
                            int L, int NB, float* __restrict__ out) {
    __shared__ float red[256];
    float s = 0.0f;
    for (int l = threadIdx.x; l < L; l += 256) {
        float m = __builtin_inff();
        for (int b = 0; b < NB; ++b) m = fminf(m, state[b * L + l]);
        for (int b = 0; b < NB; ++b) m = fminf(m, alphas[b * L + l]);
        s += m;
    }
    red[threadIdx.x] = s; __syncthreads();
    for (int off = 128; off; off >>= 1) {
        if (threadIdx.x < off) red[threadIdx.x] += red[threadIdx.x + off];
        __syncthreads();
    }
    if (threadIdx.x == 0) *out = red[0];
}

// ---------------------------------------------------------------------------
extern "C" void kernel_launch(void* const* d_in, const int* in_sizes, int n_in,
                              void* d_out, int out_size, void* d_ws, size_t ws_size,
                              hipStream_t stream) {
    (void)in_sizes; (void)n_in; (void)out_size; (void)ws_size;

    const int B = 16, LK = 2048, LQ = 64;
    const int WLD_Q = 3073, WLD_T = 5121;

    const float* query     = (const float*)d_in[0];
    const float* key       = (const float*)d_in[1];
    const float* value     = (const float*)d_in[2];
    const int*   mask      = (const int*)  d_in[3];
    const float* q_key     = (const float*)d_in[4];
    const float* q_value   = (const float*)d_in[5];
    const int*   q_mask    = (const int*)  d_in[6];
    const float* a_state   = (const float*)d_in[7];
    const float* qa_state  = (const float*)d_in[8];
    const float* W_q       = (const float*)d_in[9];
    const float* b_q       = (const float*)d_in[10];
    const float* W_t       = (const float*)d_in[11];
    const float* b_t       = (const float*)d_in[12];
    const float* q_other   = (const float*)d_in[13];
    const float* t_other   = (const float*)d_in[14];

    // d_out layout: context[32768] | alphas[32768] | q_alphas[1024] | t_loss | q_loss
    float* out_context = (float*)d_out;
    float* out_alphas  = out_context + B * HDIM;
    float* out_qalphas = out_alphas + B * LK;
    float* out_tloss   = out_qalphas + B * LQ;
    float* out_qloss   = out_tloss + 1;

    // workspace layout
    char* ws = (char*)d_ws;
    unsigned short* Wb_q = (unsigned short*)ws;                       ws += (size_t)HDIM * HDIM * 2;
    unsigned short* Wb_t = (unsigned short*)ws;                       ws += (size_t)HDIM * HDIM * 2;
    float* bc_q   = (float*)ws;  ws += (size_t)B * HDIM * 4;
    float* bc_t   = (float*)ws;  ws += (size_t)B * HDIM * 4;
    float* q_ctx  = (float*)ws;  ws += (size_t)B * HDIM * 4;
    float* sc_q   = (float*)ws;  ws += (size_t)B * LQ * 4;
    float* sc_t   = (float*)ws;  ws += (size_t)B * LK * 4;

    const size_t SMEM = (size_t)TM * LDSS * 2 + 2 * TM * sizeof(float);
    (void)hipFuncSetAttribute((const void*)gemm_score_kernel,
                              hipFuncAttributeMaxDynamicSharedMemorySize, (int)SMEM);

    // 1. bf16 weight staging (L2-resident afterwards)
    wconv_kernel<<<HDIM, 256, 0, stream>>>(W_q, Wb_q, WLD_Q);
    wconv_kernel<<<HDIM, 256, 0, stream>>>(W_t, Wb_t, WLD_T);

    // 2. query-side rank terms + GEMM-score
    bc_kernel<<<dim3(B, HDIM / 256), 256, 0, stream>>>(
        W_q, WLD_Q, b_q, query, 1024, 2048, nullptr, 0, 0, bc_q);
    gemm_score_kernel<<<(B * LQ) / TM, 256, SMEM, stream>>>(
        q_key, Wb_q, W_q, WLD_Q, bc_q, qa_state, q_other, sc_q, LQ);

    // 3. query-side softmax, context
    softmax_kernel<<<B, 256, 0, stream>>>(sc_q, q_mask, out_qalphas, LQ);
    context_kernel<<<dim3(B, HDIM / 256), 256, 0, stream>>>(out_qalphas, q_value, q_ctx, LQ);

    // 4. text-side rank terms + GEMM-score (dominant work)
    bc_kernel<<<dim3(B, HDIM / 256), 256, 0, stream>>>(
        W_t, WLD_T, b_t, query, 1024, 2048, q_ctx, 2048, 3073, bc_t);
    gemm_score_kernel<<<(B * LK) / TM, 256, SMEM, stream>>>(
        key, Wb_t, W_t, WLD_T, bc_t, a_state, t_other, sc_t, LK);

    // 5. text-side softmax, context
    softmax_kernel<<<B, 256, 0, stream>>>(sc_t, mask, out_alphas, LK);
    context_kernel<<<dim3(B, HDIM / 256), 256, 0, stream>>>(out_alphas, value, out_context, LK);

    // 6. losses
    loss_kernel<<<1, 256, 0, stream>>>(a_state, out_alphas, LK, B, out_tloss);
    loss_kernel<<<1, 256, 0, stream>>>(qa_state, out_qalphas, LQ, B, out_qloss);
}
